// Encoder_21930103013477
// MI455X (gfx1250) — compile-verified, matrix operand order checked by reading, not run on hardware
//
#include <hip/hip_runtime.h>
#include <hip/hip_bf16.h>

// ---------------------------------------------------------------------------
// DA-RNN encoder for MI455X (gfx1250): bf16 WMMA GEMMs, fp32 state/nonlinear.
// B=128, T=128, N=32, M=128.
// Sequential kernels stage all recurrent weights into LDS once via the CDNA5
// async global->LDS path (ASYNCcnt), then run the 128-step recurrence with
// LDS-only WMMA operands.
// ---------------------------------------------------------------------------

#define Bsz 128
#define Tsz 128
#define Nsz 32
#define Msz 128
#define G4M 512   // 4*M

typedef __attribute__((ext_vector_type(16))) __bf16 v16bf;
typedef __attribute__((ext_vector_type(8)))  float  v8f;

struct P {
    // inputs (setup_inputs order)
    const float *x, *Wih0, *Whh0, *bih0, *bhh0, *We, *Ue, *Ve, *Wg, *bg, *Wa, *ba;
    const float *Wih1, *Whh1, *bih1, *bhh1, *Wih2, *Whh2, *bih2, *bhh2;
    // output [B,T,2M]
    float* out;
    // workspace
    float*          ux;     // [B*T, M] fp32
    unsigned short* sa_bf;  // [B*T, N] bf16 (self attention output)
    unsigned short* ia_bf;  // [B*T, N] bf16 (input attention output)
    unsigned short* wf0;    // [32 ntile][5 kc][512]  B-frags of [Wih0|Whh0]^T
    unsigned short* wf1;
    unsigned short* wf2;
    unsigned short* wef;    // [8][8][512]  B-frags of We^T  (K=256, N=128)
    unsigned short* uef;    // [8][1][512]  B-frags of Ue^T  (K=32,  N=128)
    unsigned short* wgf;    // [8][1][512]  B-frags of Wg^T
    unsigned short* waf;    // [2][4][512]  B-frags of Wa^T  (K=128, N=32)
    float *b0, *b1, *b2;    // combined biases bih+bhh [512]
};

// round-to-nearest-even float -> bf16 bits
__device__ __forceinline__ unsigned short f2bf(float f) {
    unsigned u = __float_as_uint(f);
    u += 0x7FFFu + ((u >> 16) & 1u);
    return (unsigned short)(u >> 16);
}
__device__ __forceinline__ float sigmoidf_(float v) { return 1.f / (1.f + __expf(-v)); }

// WMMA 16x16x32 bf16 fragment coordinate map (ISA 7.12.2, 16-bit A layout;
// B fragment uses the same map with row == N column).
//  lane in [0,32), slot in [0,16) -> (row, k) with k in [0,32)
__device__ __forceinline__ void frag_coord(int lane, int slot, int& row, int& k) {
    row = lane & 15;
    int v = slot >> 1, e = slot & 1;
    k = ((v & 3) << 1) + e + ((v >> 2) << 4) + (((lane >> 4) & 1) << 3);
}

__device__ __forceinline__ v8f wmma_bf16(const unsigned short* a_ptr,
                                         const unsigned short* b_ptr, v8f acc) {
    v16bf a = *(const v16bf*)a_ptr;
    v16bf b = *(const v16bf*)b_ptr;
    return __builtin_amdgcn_wmma_f32_16x16x32_bf16(false, a, false, b,
                                                   (short)0, acc, false, false);
}

// CDNA5 async global->LDS copy (per-lane 16B), tracked with ASYNCcnt.
// Generic shared pointers carry the LDS byte offset in their low 32 bits
// (flat aperture layout, ISA 10.2), so truncation yields the DS address.
__device__ __forceinline__ void async_stage_b128(const void* gsrc, void* ldst) {
    unsigned lds_off = (unsigned)(uintptr_t)ldst;
    asm volatile("global_load_async_to_lds_b128 %0, %1, off"
                 :: "v"(lds_off), "v"(gsrc) : "memory");
}
__device__ __forceinline__ void wait_async_all() {
    asm volatile("s_wait_asynccnt 0x0" ::: "memory");
}

// ---------------------------------------------------------------------------
// Kernel 1: pre-swizzle all weights into bf16 WMMA B-fragments + fused biases
// ---------------------------------------------------------------------------
__global__ void k_prep(P p) {
    int g0 = blockIdx.x * blockDim.x + threadIdx.x;
    int gs = gridDim.x * blockDim.x;

    // [Wih|Whh]^T fragments for the three LSTMs: 32 ntiles x 5 kchunks x 512
    for (int idx = g0; idx < 32 * 5 * 512; idx += gs) {
        int nt = idx / 2560, rem = idx % 2560, kc = rem >> 9, s = rem & 511;
        int lane = s >> 4, slot = s & 15, nl, kl;
        frag_coord(lane, slot, nl, kl);
        int n = nt * 16 + nl, k = kc * 32 + kl;   // n: gate col, k in [0,160)
        float v0 = (k < Nsz) ? p.Wih0[n * Nsz + k] : p.Whh0[n * Msz + (k - Nsz)];
        float v1 = (k < Nsz) ? p.Wih1[n * Nsz + k] : p.Whh1[n * Msz + (k - Nsz)];
        float v2 = (k < Nsz) ? p.Wih2[n * Nsz + k] : p.Whh2[n * Msz + (k - Nsz)];
        p.wf0[idx] = f2bf(v0); p.wf1[idx] = f2bf(v1); p.wf2[idx] = f2bf(v2);
    }
    // We^T fragments: 8 ntiles x 8 kchunks  (wehs = [h|c] @ We^T)
    for (int idx = g0; idx < 8 * 8 * 512; idx += gs) {
        int nt = idx / 4096, rem = idx % 4096, kc = rem >> 9, s = rem & 511;
        int lane = s >> 4, slot = s & 15, nl, kl;
        frag_coord(lane, slot, nl, kl);
        int n = nt * 16 + nl, k = kc * 32 + kl;   // n<128, k<256
        p.wef[idx] = f2bf(p.We[n * (2 * Msz) + k]);
    }
    // Ue^T and Wg^T fragments: 8 ntiles x 1 kchunk
    for (int idx = g0; idx < 8 * 512; idx += gs) {
        int nt = idx >> 9, s = idx & 511;
        int lane = s >> 4, slot = s & 15, nl, kl;
        frag_coord(lane, slot, nl, kl);
        int n = nt * 16 + nl;
        p.uef[idx] = f2bf(p.Ue[n * Nsz + kl]);
        p.wgf[idx] = f2bf(p.Wg[n * Nsz + kl]);
    }
    // Wa^T fragments: 2 ntiles x 4 kchunks
    for (int idx = g0; idx < 2 * 4 * 512; idx += gs) {
        int nt = idx / 2048, rem = idx % 2048, kc = rem >> 9, s = rem & 511;
        int lane = s >> 4, slot = s & 15, nl, kl;
        frag_coord(lane, slot, nl, kl);
        int n = nt * 16 + nl, k = kc * 32 + kl;   // n<32, k<128
        p.waf[idx] = f2bf(p.Wa[n * Msz + k]);
    }
    // fused biases
    for (int idx = g0; idx < G4M; idx += gs) {
        p.b0[idx] = p.bih0[idx] + p.bhh0[idx];
        p.b1[idx] = p.bih1[idx] + p.bhh1[idx];
        p.b2[idx] = p.bih2[idx] + p.bhh2[idx];
    }
}

// ---------------------------------------------------------------------------
// Kernel 2: ux[b,t,m] = x @ Ue^T   ([16384,32] @ [32,128]) -- pure WMMA
// grid: 1024 blocks (16 rows each), 256 threads (8 waves = 8 N-tiles)
// ---------------------------------------------------------------------------
__global__ void k_ux(P p) {
    __shared__ __align__(32) unsigned short xfrag[512];
    int tid = threadIdx.x, lane = tid & 31, w = tid >> 5;
    int rowbase = blockIdx.x * 16;   // flattened b*T+t row

    for (int s = tid; s < 512; s += 256) {
        int l = s >> 4, sl = s & 15, row, k;
        frag_coord(l, sl, row, k);
        xfrag[s] = f2bf(p.x[(rowbase + row) * Nsz + k]);
    }
    __syncthreads();

    v8f acc = {};
    acc = wmma_bf16(&xfrag[lane * 16], &p.uef[w * 512 + lane * 16], acc);
    int n = lane & 15, mh = (lane >> 4) << 3;
    #pragma unroll
    for (int r = 0; r < 8; ++r)
        p.ux[(rowbase + r + mh) * Msz + w * 16 + n] = acc[r];
}

// ---------------------------------------------------------------------------
// Kernel 3: self attention  a = sigmoid(tanh(x@Wg^T+bg)@Wa^T + ba);  sa = a*x
// ---------------------------------------------------------------------------
__global__ void k_selfatt(P p) {
    __shared__ __align__(32) unsigned short xfrag[512];
    __shared__ __align__(32) unsigned short gfrag[4 * 512];
    __shared__ float gL[16 * Msz];
    int tid = threadIdx.x, lane = tid & 31, w = tid >> 5;
    int rowbase = blockIdx.x * 16;

    for (int s = tid; s < 512; s += 256) {
        int l = s >> 4, sl = s & 15, row, k;
        frag_coord(l, sl, row, k);
        xfrag[s] = f2bf(p.x[(rowbase + row) * Nsz + k]);
    }
    __syncthreads();

    // g = tanh(x @ Wg^T + bg): 8 N-tiles, one per wave
    {
        float bv = p.bg[w * 16 + (lane & 15)];
        v8f acc;
        #pragma unroll
        for (int r = 0; r < 8; ++r) acc[r] = bv;
        acc = wmma_bf16(&xfrag[lane * 16], &p.wgf[w * 512 + lane * 16], acc);
        int n = lane & 15, mh = (lane >> 4) << 3;
        #pragma unroll
        for (int r = 0; r < 8; ++r)
            gL[(r + mh) * Msz + w * 16 + n] = tanhf(acc[r]);
    }
    __syncthreads();

    // re-stage g as A fragments (K=128 -> 4 kchunks)
    for (int s = tid; s < 4 * 512; s += 256) {
        int kc = s >> 9, r = s & 511, l = r >> 4, sl = r & 15, row, kl;
        frag_coord(l, sl, row, kl);
        gfrag[s] = f2bf(gL[row * Msz + kc * 32 + kl]);
    }
    __syncthreads();

    // a = sigmoid(g @ Wa^T + ba): N=32 -> 2 tiles on waves 0..1
    if (w < 2) {
        float bv = p.ba[w * 16 + (lane & 15)];
        v8f acc;
        #pragma unroll
        for (int r = 0; r < 8; ++r) acc[r] = bv;
        #pragma unroll
        for (int kc = 0; kc < 4; ++kc)
            acc = wmma_bf16(&gfrag[kc * 512 + lane * 16],
                            &p.waf[(w * 4 + kc) * 512 + lane * 16], acc);
        int n = lane & 15, mh = (lane >> 4) << 3;
        #pragma unroll
        for (int r = 0; r < 8; ++r) {
            int m = r + mh, col = w * 16 + n;
            float av = sigmoidf_(acc[r]);
            float xv = p.x[(rowbase + m) * Nsz + col];
            p.sa_bf[(rowbase + m) * Nsz + col] = f2bf(av * xv);
        }
    }
}

// ---------------------------------------------------------------------------
// Kernel 4: LSTM0 + input attention, sequential over T.
// grid: 8 blocks (16 batch rows each), 256 threads (8 waves).
// All recurrent weights async-staged into LDS once (~302 KB total LDS).
// ---------------------------------------------------------------------------
__global__ void k_lstm0_attn(P p) {
    __shared__ float hL[16 * Msz], cL[16 * Msz];
    __shared__ float wehsL[16 * Msz];
    __shared__ float eL[16 * Tsz];
    __shared__ float gatesL[16 * G4M];
    __shared__ float veL[Msz];
    __shared__ __align__(32) unsigned short hcfrag[8 * 512];
    __shared__ __align__(32) unsigned short zfrag[5 * 512];
    __shared__ __align__(32) unsigned short wf0L[32 * 5 * 512];  // 160 KB
    __shared__ __align__(32) unsigned short wefL[8 * 8 * 512];   //  64 KB

    int tid = threadIdx.x, lane = tid & 31, w = tid >> 5;
    int b0r = blockIdx.x * 16;    // batch row base

    // async-stage weight fragments into LDS (16B per lane per issue)
    for (int i = tid * 8; i < 32 * 5 * 512; i += 256 * 8)
        async_stage_b128(p.wf0 + i, &wf0L[i]);
    for (int i = tid * 8; i < 8 * 8 * 512; i += 256 * 8)
        async_stage_b128(p.wef + i, &wefL[i]);
    wait_async_all();

    for (int u = tid; u < 16 * Msz; u += 256) { hL[u] = 0.f; cL[u] = 0.f; }
    for (int u = tid; u < Msz; u += 256) veL[u] = p.Ve[u];
    __syncthreads();

    for (int t = 0; t < Tsz; ++t) {
        // --- [h|c] -> A fragments (K=256, 8 chunks) ---
        for (int s = tid; s < 8 * 512; s += 256) {
            int kc = s >> 9, r = s & 511, l = r >> 4, sl = r & 15, row, kl;
            frag_coord(l, sl, row, kl);
            int k = kc * 32 + kl;
            float v = (k < Msz) ? hL[row * Msz + k] : cL[row * Msz + (k - Msz)];
            hcfrag[s] = f2bf(v);
        }
        __syncthreads();

        // --- wehs = [h|c] @ We^T : 8 N-tiles, K=256 ---
        {
            v8f acc = {};
            #pragma unroll
            for (int kc = 0; kc < 8; ++kc)
                acc = wmma_bf16(&hcfrag[kc * 512 + lane * 16],
                                &wefL[(w * 8 + kc) * 512 + lane * 16], acc);
            int n = lane & 15, mh = (lane >> 4) << 3;
            #pragma unroll
            for (int r = 0; r < 8; ++r)
                wehsL[(r + mh) * Msz + w * 16 + n] = acc[r];
        }
        __syncthreads();

        // --- e[row,tt] = sum_m Ve[m]*tanh(ux[b,tt,m] + wehs[row,m]) ---
        for (int pr = tid; pr < 16 * Tsz; pr += 256) {
            int row = pr >> 7, tt = pr & 127;
            const float* uxp = &p.ux[((b0r + row) * Tsz + tt) * Msz];
            const float* wp = &wehsL[row * Msz];
            float s = 0.f;
            #pragma unroll 4
            for (int m = 0; m < Msz; ++m)
                s += veL[m] * tanhf(uxp[m] + wp[m]);
            eL[pr] = s;
        }
        __syncthreads();

        // --- softmax over tt per row (16 rows) ---
        if (tid < 16) {
            float* ep = &eL[tid * Tsz];
            float mx = -3.0e38f;
            for (int tt = 0; tt < Tsz; ++tt) mx = fmaxf(mx, ep[tt]);
            float sum = 0.f;
            for (int tt = 0; tt < Tsz; ++tt) { float ev = __expf(ep[tt] - mx); ep[tt] = ev; sum += ev; }
            float inv = 1.f / sum;
            for (int tt = 0; tt < Tsz; ++tt) ep[tt] *= inv;
        }
        __syncthreads();

        // --- ia_t[row,n] = sum_tt a[row,tt] * x[b,tt,n]  -> bf16 stream ---
        for (int o = tid; o < 16 * Nsz; o += 256) {
            int row = o >> 5, n = o & 31;
            const float* ap = &eL[row * Tsz];
            const float* xp = &p.x[(b0r + row) * Tsz * Nsz + n];
            float s = 0.f;
            #pragma unroll 4
            for (int tt = 0; tt < Tsz; ++tt) s += ap[tt] * xp[tt * Nsz];
            p.ia_bf[((b0r + row) * Tsz + t) * Nsz + n] = f2bf(s);
        }
        __syncthreads();

        // --- z = [x_t | h] -> A fragments (K=160, 5 chunks) ---
        for (int s = tid; s < 5 * 512; s += 256) {
            int kc = s >> 9, r = s & 511, l = r >> 4, sl = r & 15, row, kl;
            frag_coord(l, sl, row, kl);
            int k = kc * 32 + kl;
            float v = (k < Nsz) ? p.x[((b0r + row) * Tsz + t) * Nsz + k]
                                : hL[row * Msz + (k - Nsz)];
            zfrag[s] = f2bf(v);
        }
        __syncthreads();

        // --- gates = z @ [Wih0|Whh0]^T + b : 32 N-tiles, 4 per wave ---
        for (int nt = w; nt < 32; nt += 8) {
            float bv = p.b0[nt * 16 + (lane & 15)];
            v8f acc;
            #pragma unroll
            for (int r = 0; r < 8; ++r) acc[r] = bv;
            #pragma unroll
            for (int kc = 0; kc < 5; ++kc)
                acc = wmma_bf16(&zfrag[kc * 512 + lane * 16],
                                &wf0L[(nt * 5 + kc) * 512 + lane * 16], acc);
            int n = lane & 15, mh = (lane >> 4) << 3;
            #pragma unroll
            for (int r = 0; r < 8; ++r)
                gatesL[(r + mh) * G4M + nt * 16 + n] = acc[r];
        }
        __syncthreads();

        // --- cell update (i,f,g,o) ---
        for (int u = tid; u < 16 * Msz; u += 256) {
            int row = u >> 7, j = u & 127;
            const float* gp = &gatesL[row * G4M];
            float gi = gp[j], gf = gp[Msz + j], gg = gp[2 * Msz + j], go = gp[3 * Msz + j];
            float cc = sigmoidf_(gf) * cL[u] + sigmoidf_(gi) * tanhf(gg);
            cL[u] = cc;
            hL[u] = sigmoidf_(go) * tanhf(cc);
        }
        __syncthreads();
    }
}

// ---------------------------------------------------------------------------
// Kernel 5: LSTM1 (input attention) and LSTM2 (self attention), fused launch.
// grid: (8, 2); blockIdx.y selects the sequence. Writes d_out directly.
// Recurrent weights async-staged into LDS once (~213 KB total LDS).
// ---------------------------------------------------------------------------
__global__ void k_lstm12(P p) {
    __shared__ float hL[16 * Msz], cL[16 * Msz];
    __shared__ float gatesL[16 * G4M];
    __shared__ __align__(32) unsigned short zfrag[5 * 512];
    __shared__ __align__(32) unsigned short wfL[32 * 5 * 512];   // 160 KB

    int tid = threadIdx.x, lane = tid & 31, w = tid >> 5;
    int b0r = blockIdx.x * 16;
    const unsigned short* inb  = blockIdx.y ? p.sa_bf : p.ia_bf;
    const unsigned short* wf   = blockIdx.y ? p.wf2 : p.wf1;
    const float*          bias = blockIdx.y ? p.b2 : p.b1;
    int outbase = blockIdx.y ? Msz : 0;

    for (int i = tid * 8; i < 32 * 5 * 512; i += 256 * 8)
        async_stage_b128(wf + i, &wfL[i]);
    wait_async_all();

    for (int u = tid; u < 16 * Msz; u += 256) { hL[u] = 0.f; cL[u] = 0.f; }
    __syncthreads();

    for (int t = 0; t < Tsz; ++t) {
        // z = [in_t | h] fragments (chunk0 is already bf16)
        for (int s = tid; s < 5 * 512; s += 256) {
            int kc = s >> 9, r = s & 511, l = r >> 4, sl = r & 15, row, kl;
            frag_coord(l, sl, row, kl);
            int k = kc * 32 + kl;
            unsigned short v = (k < Nsz)
                ? inb[((b0r + row) * Tsz + t) * Nsz + k]
                : f2bf(hL[row * Msz + (k - Nsz)]);
            zfrag[s] = v;
        }
        __syncthreads();

        for (int nt = w; nt < 32; nt += 8) {
            float bv = bias[nt * 16 + (lane & 15)];
            v8f acc;
            #pragma unroll
            for (int r = 0; r < 8; ++r) acc[r] = bv;
            #pragma unroll
            for (int kc = 0; kc < 5; ++kc)
                acc = wmma_bf16(&zfrag[kc * 512 + lane * 16],
                                &wfL[(nt * 5 + kc) * 512 + lane * 16], acc);
            int n = lane & 15, mh = (lane >> 4) << 3;
            #pragma unroll
            for (int r = 0; r < 8; ++r)
                gatesL[(r + mh) * G4M + nt * 16 + n] = acc[r];
        }
        __syncthreads();

        for (int u = tid; u < 16 * Msz; u += 256) {
            int row = u >> 7, j = u & 127;
            const float* gp = &gatesL[row * G4M];
            float gi = gp[j], gf = gp[Msz + j], gg = gp[2 * Msz + j], go = gp[3 * Msz + j];
            float cc = sigmoidf_(gf) * cL[u] + sigmoidf_(gi) * tanhf(gg);
            cL[u] = cc;
            float hv = sigmoidf_(go) * tanhf(cc);
            hL[u] = hv;
            p.out[((b0r + row) * Tsz + t) * (2 * Msz) + outbase + j] = hv;
        }
        __syncthreads();
    }
}

// ---------------------------------------------------------------------------
extern "C" void kernel_launch(void* const* d_in, const int* in_sizes, int n_in,
                              void* d_out, int out_size, void* d_ws, size_t ws_size,
                              hipStream_t stream) {
    (void)in_sizes; (void)n_in; (void)out_size; (void)ws_size;
    P p;
    p.x    = (const float*)d_in[0];
    p.Wih0 = (const float*)d_in[1];  p.Whh0 = (const float*)d_in[2];
    p.bih0 = (const float*)d_in[3];  p.bhh0 = (const float*)d_in[4];
    p.We   = (const float*)d_in[5];  p.Ue   = (const float*)d_in[6];
    p.Ve   = (const float*)d_in[7];
    p.Wg   = (const float*)d_in[8];  p.bg   = (const float*)d_in[9];
    p.Wa   = (const float*)d_in[10]; p.ba   = (const float*)d_in[11];
    p.Wih1 = (const float*)d_in[12]; p.Whh1 = (const float*)d_in[13];
    p.bih1 = (const float*)d_in[14]; p.bhh1 = (const float*)d_in[15];
    p.Wih2 = (const float*)d_in[16]; p.Whh2 = (const float*)d_in[17];
    p.bih2 = (const float*)d_in[18]; p.bhh2 = (const float*)d_in[19];
    p.out  = (float*)d_out;

    char* ws = (char*)d_ws;
    size_t off = 0;
    auto bump = [&](size_t bytes) -> void* {
        void* ptr = ws + off;
        off += (bytes + 255) & ~size_t(255);
        return ptr;
    };
    p.ux    = (float*)bump((size_t)Bsz * Tsz * Msz * sizeof(float));
    p.sa_bf = (unsigned short*)bump((size_t)Bsz * Tsz * Nsz * 2);
    p.ia_bf = (unsigned short*)bump((size_t)Bsz * Tsz * Nsz * 2);
    p.wf0   = (unsigned short*)bump(32 * 5 * 512 * 2);
    p.wf1   = (unsigned short*)bump(32 * 5 * 512 * 2);
    p.wf2   = (unsigned short*)bump(32 * 5 * 512 * 2);
    p.wef   = (unsigned short*)bump(8 * 8 * 512 * 2);
    p.uef   = (unsigned short*)bump(8 * 512 * 2);
    p.wgf   = (unsigned short*)bump(8 * 512 * 2);
    p.waf   = (unsigned short*)bump(2 * 4 * 512 * 2);
    p.b0    = (float*)bump(G4M * sizeof(float));
    p.b1    = (float*)bump(G4M * sizeof(float));
    p.b2    = (float*)bump(G4M * sizeof(float));

    int rowTiles = (Bsz * Tsz) / 16;                 // 1024
    k_prep      <<<128, 256, 0, stream>>>(p);
    k_ux        <<<rowTiles, 256, 0, stream>>>(p);
    k_selfatt   <<<rowTiles, 256, 0, stream>>>(p);
    k_lstm0_attn<<<Bsz / 16, 256, 0, stream>>>(p);
    k_lstm12    <<<dim3(Bsz / 16, 2), 256, 0, stream>>>(p);
}
